// WD_LSTM_GAT_12008728559869
// MI455X (gfx1250) — matrix operand order repaired and checked by reading, not compiled, not run on hardware
//
#include <hip/hip_runtime.h>
#include <hip/hip_bf16.h>
#include <cstdint>
#include <cstddef>

typedef __attribute__((ext_vector_type(16))) _Float16 v16h;
typedef __attribute__((ext_vector_type(8)))  _Float16 h8;
typedef __attribute__((ext_vector_type(4)))  _Float16 h4;
typedef __attribute__((ext_vector_type(4)))  float    f4;
typedef __attribute__((ext_vector_type(8)))  float    v8f;

#define HB    32            // batch B
#define TT    35            // T
#define GG    (HB * TT)     // 1120 rows
#define DD    300           // D
#define NNODE 32            // N nodes per graph
#define EE    64            // E edges
#define HHEAD 4             // H heads
#define CCH   75            // C = D/H
#define UU    1024          // U hidden
#define GATE  4096          // 4*U
#define VGL   50000         // output vocab
#define SIGP  608           // 600 padded to K-multiple of 32
#define GWN   304           // gat_W N padded
#define GWK   320           // gat_W K padded

// ---------------------------------------------------------------------------
// WMMA helper: D = A(16x32 f16) * B(32x16 f16) + C(16x16 f32)
// ---------------------------------------------------------------------------
__device__ __forceinline__ v8f wmma16(v16h a, v16h b, v8f c) {
  return __builtin_amdgcn_wmma_f32_16x16x32_f16(
      /*neg_a=*/false, a, /*neg_b=*/false, b,
      /*c_mod=*/(short)0, c, /*reuse_a=*/false, /*reuse_b=*/false);
}

// A fragment (rows m0..m0+15 of row-major f16, K slab [k0,k0+32)).
// ISA layout: lane grp g, VGPRs 0-3 hold K=[g*8, g*8+8), VGPRs 4-7 hold
// K=[16+g*8, 16+g*8+8)  ->  two contiguous 16-byte loads per lane.
// Requires: rstride, base, k0 all multiples of 8 elements (16B aligned).
__device__ __forceinline__ v16h load_a(const _Float16* p, int base, int rstride,
                                       int m0, int k0, int lane) {
  int grp = lane >> 4, m = lane & 15;
  const _Float16* row = p + (size_t)base + (size_t)(m0 + m) * rstride + k0;
  h8 lo = *(const h8*)(row + grp * 8);
  h8 hi = *(const h8*)(row + 16 + grp * 8);
  v16h a;
#pragma unroll
  for (int i = 0; i < 8; ++i) { a[i] = lo[i]; a[8 + i] = hi[i]; }
  return a;
}

// B fragment from row-major (N,K) f16 weight: B_logical[k][n] = W[n*ldk + k].
// ISA layout: lane grp g holds K=[k0+g*8, k0+g*8+8) then
// K=[k0+16+g*8, k0+16+g*8+8) -> two 16-byte loads per lane.
__device__ __forceinline__ v16h load_b(const _Float16* W, int n0, int ldk,
                                       int k0, int lane) {
  int grp = lane >> 4, n = lane & 15;
  const _Float16* row = W + (size_t)(n0 + n) * ldk + k0;
  h8 lo = *(const h8*)(row + grp * 8);
  h8 hi = *(const h8*)(row + 16 + grp * 8);
  v16h b;
#pragma unroll
  for (int i = 0; i < 8; ++i) { b[i] = lo[i]; b[8 + i] = hi[i]; }
  return b;
}

// ---------------------------------------------------------------------------
// weight conversion kernels
// ---------------------------------------------------------------------------
__global__ void f32_to_f16_kernel(const float* __restrict__ s,
                                  _Float16* __restrict__ d, int n) {
  int i = blockIdx.x * blockDim.x + threadIdx.x;
  if (i < n) d[i] = (_Float16)s[i];
}

// W_ih0 (4096 x 600) -> padded f16 (4096 x 608)
__global__ void wih0_to_f16_kernel(const float* __restrict__ s,
                                   _Float16* __restrict__ d) {
  int i = blockIdx.x * blockDim.x + threadIdx.x;
  if (i >= GATE * SIGP) return;
  int n = i / SIGP, k = i - n * SIGP;
  d[i] = (_Float16)((k < 600) ? s[n * 600 + k] : 0.f);
}

// gat_W (K=300 x N=300, K-major) -> transposed padded f16 (N=304 x K=320)
__global__ void gatW_to_f16T_kernel(const float* __restrict__ s,
                                    _Float16* __restrict__ d) {
  int i = blockIdx.x * blockDim.x + threadIdx.x;
  if (i >= GWN * GWK) return;
  int n = i / GWK, k = i - n * GWK;
  d[i] = (_Float16)((n < DD && k < DD) ? s[k * DD + n] : 0.f);
}

// ---------------------------------------------------------------------------
// GAT: one workgroup per graph g. h = X[idx] @ gat_W via WMMA, masked node-0
// attention, writes sig[g] = [curr_emb(300) | node0(300) | pad(8)] f16.
// ---------------------------------------------------------------------------
__global__ void gat_kernel(const int* __restrict__ xidx, const int* __restrict__ eidx,
                           const float* __restrict__ X,
                           const _Float16* __restrict__ gatWT,
                           const float* __restrict__ attS, const float* __restrict__ attD,
                           const float* __restrict__ gbias, _Float16* __restrict__ sig) {
  __shared__ __align__(16) _Float16 sXh[NNODE * GWK];   // 20 KB
  __shared__ __align__(16) float    sH[NNODE * GWN];    // 38.9 KB
  __shared__ float sAs[NNODE * HHEAD], sAd[NNODE * HHEAD];
  __shared__ float sAlpha[HHEAD * NNODE];
  __shared__ int   sMask[NNODE];

  int g = blockIdx.x, tid = threadIdx.x;
  int lane = tid & 31, wave = tid >> 5;
  const int* idx = xidx + g * NNODE;

  // gather node features (f32x4 loads -> f16x4 stores), zero-pad K 300..319
  for (int i = tid; i < NNODE * (DD / 4); i += 256) {
    int n = i / (DD / 4), kk = i - n * (DD / 4);
    f4 v = *(const f4*)(X + (size_t)idx[n] * DD + kk * 4);
    h4 o;
#pragma unroll
    for (int j = 0; j < 4; ++j) o[j] = (_Float16)v[j];
    *(h4*)(sXh + n * GWK + kk * 4) = o;
  }
  for (int i = tid; i < NNODE * 20; i += 256) {
    int n = i / 20;
    sXh[n * GWK + 300 + (i - n * 20)] = (_Float16)0.f;
  }
  if (tid < NNODE) sMask[tid] = (tid == 0) ? 1 : 0;     // self-loop on node 0
  __syncthreads();

  // h(32x304) = x(32x320) @ W(320x304) : 2 M-tiles x 19 N-tiles, 10 K-steps
  for (int tj = wave; tj < 38; tj += 8) {
    int mt = tj / 19, nt = tj - mt * 19;
    int m0 = mt * 16, n0 = nt * 16;
    v8f acc = {};
    for (int ks = 0; ks < GWK / 32; ++ks) {
      v16h a = load_a(sXh, 0, GWK, m0, ks * 32, lane);
      v16h b = load_b(gatWT, n0, GWK, ks * 32, lane);
      acc = wmma16(a, b, acc);
    }
    int n = lane & 15, grp = lane >> 4;
#pragma unroll
    for (int r = 0; r < 8; ++r) {
      int m = m0 + grp * 8 + r;
      sH[m * GWN + n0 + n] = acc[r];
    }
  }
  __syncthreads();

  // attention scalars a_s, a_d (32 nodes x 4 heads)
  if (tid < NNODE * HHEAD) {
    int n = tid >> 2, hd = tid & 3;
    float as = 0.f, ad = 0.f;
    for (int c = 0; c < CCH; ++c) {
      float v = sH[n * GWN + hd * CCH + c];
      as += v * attS[hd * CCH + c];
      ad += v * attD[hd * CCH + c];
    }
    sAs[n * HHEAD + hd] = as;
    sAd[n * HHEAD + hd] = ad;
  }
  __syncthreads();

  // adjacency row 0: edges with dst == 0
  if (tid < EE) {
    int src = eidx[g * 2 * EE + tid];
    int dst = eidx[g * 2 * EE + EE + tid];
    if (dst == 0) sMask[src] = 1;
  }
  __syncthreads();

  // masked softmax for query node 0 (one thread per head)
  if (tid < HHEAD) {
    int hd = tid;
    float ad0 = sAd[hd];
    float mx = -3.4e38f;
    for (int j = 0; j < NNODE; ++j) {
      float v = ad0 + sAs[j * HHEAD + hd];
      v = (v > 0.f) ? v : 0.2f * v;                     // leaky_relu(0.2)
      v = sMask[j] ? v : -1.0e9f;
      sAlpha[hd * NNODE + j] = v;
      mx = fmaxf(mx, v);
    }
    float s = 0.f;
    for (int j = 0; j < NNODE; ++j) {
      float e = expf(sAlpha[hd * NNODE + j] - mx);
      sAlpha[hd * NNODE + j] = e;
      s += e;
    }
    float inv = 1.f / s;
    for (int j = 0; j < NNODE; ++j) sAlpha[hd * NNODE + j] *= inv;
  }
  __syncthreads();

  // node0 output + bias, curr_emb = X[idx[0]], zero the 8-wide pad
  if (tid < DD) {
    int hd = tid / CCH;
    float acc = 0.f;
    for (int j = 0; j < NNODE; ++j) acc += sAlpha[hd * NNODE + j] * sH[j * GWN + tid];
    sig[(size_t)g * SIGP + 300 + tid] = (_Float16)(acc + gbias[tid]);
    sig[(size_t)g * SIGP + tid]       = (_Float16)X[(size_t)idx[0] * DD + tid];
  }
  if (tid < 8) sig[(size_t)g * SIGP + 600 + tid] = (_Float16)0.f;
}

// ---------------------------------------------------------------------------
// LSTM gate GEMM: gates(32x4096) = x_t @ Wx^T + h_prev @ Wh^T + (b_ih+b_hh)
// 512 waves (64 blocks x 8 waves), one 16x16 tile each. Kx is pre-padded.
// ---------------------------------------------------------------------------
__global__ void lstm_gemm_kernel(const _Float16* __restrict__ xin, int xbase,
                                 int xrstride, int Kx,
                                 const _Float16* __restrict__ Wx,
                                 const _Float16* __restrict__ hprev,
                                 const _Float16* __restrict__ Wh,
                                 const float* __restrict__ bi,
                                 const float* __restrict__ bh,
                                 float* __restrict__ gates, int has_h) {
  int tid = threadIdx.x, lane = tid & 31;
  int w = (blockIdx.x * blockDim.x + tid) >> 5;        // 0..511
  int mt = w & 1, nt = w >> 1;                         // 2 M-tiles x 256 N-tiles
  int m0 = mt * 16, n0 = nt * 16;

  v8f acc = {};
  for (int ks = 0; ks < (Kx >> 5); ++ks) {
    v16h a = load_a(xin, xbase, xrstride, m0, ks * 32, lane);
    v16h b = load_b(Wx, n0, Kx, ks * 32, lane);
    acc = wmma16(a, b, acc);
  }
  if (has_h) {
    for (int ks = 0; ks < UU / 32; ++ks) {
      v16h a = load_a(hprev, 0, UU, m0, ks * 32, lane);
      v16h b = load_b(Wh, n0, UU, ks * 32, lane);
      acc = wmma16(a, b, acc);
    }
  }
  int n = lane & 15, grp = lane >> 4;
  float bias = bi[n0 + n] + bh[n0 + n];
#pragma unroll
  for (int r = 0; r < 8; ++r) {
    int m = m0 + grp * 8 + r;
    gates[(size_t)m * GATE + n0 + n] = acc[r] + bias;
  }
}

// LSTM pointwise cell update
__global__ void lstm_point_kernel(const float* __restrict__ gates,
                                  float* __restrict__ cbuf,
                                  _Float16* __restrict__ hbuf,
                                  _Float16* __restrict__ ys, int t, int first) {
  int i = blockIdx.x * blockDim.x + threadIdx.x;
  if (i >= HB * UU) return;
  int b = i >> 10, u = i & 1023;
  const float* gp = gates + (size_t)b * GATE;
  float ig = gp[u], fg = gp[UU + u], gg = gp[2 * UU + u], og = gp[3 * UU + u];
  float cp = first ? 0.f : cbuf[i];
  float si = 1.f / (1.f + expf(-ig));
  float sf = 1.f / (1.f + expf(-fg));
  float so = 1.f / (1.f + expf(-og));
  float c = sf * cp + si * tanhf(gg);
  float h = so * tanhf(c);
  cbuf[i] = c;
  hbuf[i] = (_Float16)h;
  ys[((size_t)b * TT + t) * UU + u] = (_Float16)h;
}

// ---------------------------------------------------------------------------
// logits = lstm_out(1120x1024) @ lin_W^T(1024x50000) + lin_b
// One workgroup per 16-column strip; strip converted f32->f16 into LDS once,
// 8 waves sweep 70 M-tiles x 32 K-steps of WMMA. A prefetched one tile ahead.
// ---------------------------------------------------------------------------
__global__ void logits_kernel(const _Float16* __restrict__ Ah,
                              const float* __restrict__ linW,
                              const float* __restrict__ linb,
                              float* __restrict__ out) {
  __shared__ __align__(16) _Float16 sB[16 * UU];       // 32 KB
  __shared__ float sBias[16];
  int tid = threadIdx.x, lane = tid & 31, wave = tid >> 5;
  int n0 = blockIdx.x * 16;

  // f32x4 global loads -> f16x4 ds stores
  for (int i = tid; i < 16 * (UU / 4); i += 256) {
    int n = i >> 8, kk = i & 255;                      // 256 vec4 per row
    f4 v = *(const f4*)(linW + (size_t)(n0 + n) * UU + kk * 4);
    h4 o;
#pragma unroll
    for (int j = 0; j < 4; ++j) o[j] = (_Float16)v[j];
    *(h4*)(sB + n * UU + kk * 4) = o;
  }
  if (tid < 16) sBias[tid] = linb[n0 + tid];
  __syncthreads();

  for (int mt = wave; mt < GG / 16; mt += 8) {
    int m0 = mt * 16;
    if (mt + 8 < GG / 16)                              // prefetch next A tile
      __builtin_prefetch(Ah + (size_t)(m0 + 128) * UU + (lane << 6), 0, 1);
    v8f acc = {};
    for (int ks = 0; ks < UU / 32; ++ks) {
      v16h a = load_a(Ah, 0, UU, m0, ks * 32, lane);
      v16h b = load_b(sB, 0, UU, ks * 32, lane);
      acc = wmma16(a, b, acc);
    }
    int n = lane & 15, grp = lane >> 4;
    float bias = sBias[n];
#pragma unroll
    for (int r = 0; r < 8; ++r) {
      int m = m0 + grp * 8 + r;
      out[(size_t)m * VGL + n0 + n] = acc[r] + bias;
    }
  }
}

// log_softmax over each row of 50000 (in place)
__global__ void logsoftmax_kernel(float* __restrict__ out) {
  __shared__ float red[256];
  int row = blockIdx.x, tid = threadIdx.x;
  float* p = out + (size_t)row * VGL;

  float mx = -3.4e38f;
  for (int j = tid; j < VGL; j += 256) mx = fmaxf(mx, p[j]);
  red[tid] = mx;
  __syncthreads();
  for (int s = 128; s > 0; s >>= 1) {
    if (tid < s) red[tid] = fmaxf(red[tid], red[tid + s]);
    __syncthreads();
  }
  mx = red[0];
  __syncthreads();

  float sum = 0.f;
  for (int j = tid; j < VGL; j += 256) sum += expf(p[j] - mx);
  red[tid] = sum;
  __syncthreads();
  for (int s = 128; s > 0; s >>= 1) {
    if (tid < s) red[tid] += red[tid + s];
    __syncthreads();
  }
  float lse = logf(red[0]) + mx;
  __syncthreads();

  for (int j = tid; j < VGL; j += 256) p[j] = p[j] - lse;
}

// ---------------------------------------------------------------------------
// host launch
// ---------------------------------------------------------------------------
extern "C" void kernel_launch(void* const* d_in, const int* in_sizes, int n_in,
                              void* d_out, int out_size, void* d_ws, size_t ws_size,
                              hipStream_t stream) {
  (void)in_sizes; (void)n_in; (void)out_size; (void)ws_size;
  const int*   xidx  = (const int*)d_in[0];
  const int*   eidx  = (const int*)d_in[1];
  const float* X     = (const float*)d_in[2];
  const float* gatW  = (const float*)d_in[3];
  const float* attS  = (const float*)d_in[4];
  const float* attD  = (const float*)d_in[5];
  const float* gbias = (const float*)d_in[6];
  const float* Wih0  = (const float*)d_in[7];
  const float* WihR  = (const float*)d_in[8];
  const float* Whh   = (const float*)d_in[9];
  const float* bih   = (const float*)d_in[10];
  const float* bhh   = (const float*)d_in[11];
  const float* linW  = (const float*)d_in[12];
  const float* linb  = (const float*)d_in[13];
  float* out = (float*)d_out;

  char* ws = (char*)d_ws;
  size_t off = 0;
  auto alloc = [&](size_t bytes) -> void* {
    void* p = ws + off;
    off = (off + bytes + 255) & ~(size_t)255;
    return p;
  };
  _Float16* sigF16 = (_Float16*)alloc((size_t)GG * SIGP * 2);
  _Float16* Wih0h  = (_Float16*)alloc((size_t)GATE * SIGP * 2);
  _Float16* WihRh  = (_Float16*)alloc((size_t)2 * GATE * UU * 2);
  _Float16* Whhh   = (_Float16*)alloc((size_t)3 * GATE * UU * 2);
  _Float16* gatWT  = (_Float16*)alloc((size_t)GWN * GWK * 2);
  _Float16* ysA    = (_Float16*)alloc((size_t)GG * UU * 2);
  _Float16* ysB    = (_Float16*)alloc((size_t)GG * UU * 2);
  _Float16* hbuf   = (_Float16*)alloc((size_t)HB * UU * 2);
  float*    gates  = (float*)alloc((size_t)HB * GATE * 4);
  float*    cbuf   = (float*)alloc((size_t)HB * UU * 4);

  // per-call f16 weight images (padded where K is not a multiple of 32)
  {
    int n0 = GATE * SIGP;
    wih0_to_f16_kernel<<<(n0 + 255) / 256, 256, 0, stream>>>(Wih0, Wih0h);
    int n1 = 2 * GATE * UU;
    f32_to_f16_kernel<<<(n1 + 255) / 256, 256, 0, stream>>>(WihR, WihRh, n1);
    int n2 = 3 * GATE * UU;
    f32_to_f16_kernel<<<(n2 + 255) / 256, 256, 0, stream>>>(Whh, Whhh, n2);
    int n3 = GWN * GWK;
    gatW_to_f16T_kernel<<<(n3 + 255) / 256, 256, 0, stream>>>(gatW, gatWT);
  }

  // GAT: produces padded 608-wide LSTM input signal per (b,t)
  gat_kernel<<<GG, 256, 0, stream>>>(xidx, eidx, X, gatWT, attS, attD, gbias, sigF16);

  // 3-layer LSTM, recurrence carried by stream ordering
  _Float16* ysIn  = nullptr;
  _Float16* ysOut = ysA;
  for (int l = 0; l < 3; ++l) {
    const _Float16* Wx  = (l == 0) ? Wih0h : (WihRh + (size_t)(l - 1) * GATE * UU);
    const _Float16* Wh  = Whhh + (size_t)l * GATE * UU;
    const _Float16* src = (l == 0) ? sigF16 : ysIn;
    int Kx      = (l == 0) ? SIGP : UU;
    int rstride = (l == 0) ? TT * SIGP : TT * UU;
    for (int t = 0; t < TT; ++t) {
      int xbase = (l == 0) ? t * SIGP : t * UU;
      lstm_gemm_kernel<<<64, 256, 0, stream>>>(src, xbase, rstride, Kx, Wx,
                                               hbuf, Wh, bih + l * GATE,
                                               bhh + l * GATE, gates,
                                               (t > 0) ? 1 : 0);
      lstm_point_kernel<<<(HB * UU + 255) / 256, 256, 0, stream>>>(
          gates, cbuf, hbuf, ysOut, t, (t == 0) ? 1 : 0);
    }
    ysIn  = ysOut;
    ysOut = (ysOut == ysA) ? ysB : ysA;
  }

  // final projection + log_softmax
  logits_kernel<<<VGL / 16, 256, 0, stream>>>(ysIn, linW, linb, out);
  logsoftmax_kernel<<<GG, 256, 0, stream>>>(out);
}
